// OldActionTransformer_49667001811348
// MI455X (gfx1250) — matrix operand order repaired and checked by reading, not compiled, not run on hardware
//
#include <hip/hip_runtime.h>
#include <hip/hip_bf16.h>
#include <stdint.h>

typedef __bf16 bf16;
typedef __attribute__((ext_vector_type(16))) bf16  v16bf;
typedef __attribute__((ext_vector_type(8)))  float v8f;

#define Bn   4
#define CINn 64
#define Ln   1024
#define Dn   1024
#define Hn   16
#define DHn  64
#define Sn   512
#define HIDn 2730
#define HIDP 2752
#define QKVN 3072
#define Mrow (Bn*Ln)     // 4096
#define Msty (Bn*Sn)     // 2048

// ---------------------------------------------------------------------------
// WMMA fragment helpers (layouts per CDNA5 ISA 7.12.2, wave32).
// Both loaders require kstride==1 so each lane reads contiguous 16B/32B runs.
// ---------------------------------------------------------------------------
__device__ __forceinline__ int lane_id() { return threadIdx.x & 31; }

__device__ __forceinline__ v8f zero8() {
  v8f z;
#pragma unroll
  for (int i = 0; i < 8; i++) z[i] = 0.f;
  return z;
}

// A 16x32 bf16: lanes 0-15 rows M hold K 0..7 & 16..23, lanes 16-31 K 8..15 & 24..31.
__device__ __forceinline__ v16bf load_a16(const bf16* p, int mstride) {
  int lane = lane_id();
  int m  = lane & 15;
  int kb = (lane >> 4) << 3;   // 0 or 8
  const bf16* q = p + m * mstride + kb;
  union { uint4 u[2]; v16bf v; } un;
  un.u[0] = *(const uint4*)q;         // K kb..kb+7
  un.u[1] = *(const uint4*)(q + 16);  // K kb+16..kb+23
  return un.v;
}

// B 32x16 bf16: lanes 0-15 cols N hold K 0..7 & 16..23, lanes 16-31 K 8..15 & 24..31.
// Element (k,n) at p[k + n*nstride].
__device__ __forceinline__ v16bf load_b16(const bf16* p, int nstride) {
  int lane = lane_id();
  int n  = lane & 15;
  int kb = (lane >> 4) << 3;   // 0 or 8
  const bf16* q = p + n * nstride + kb;
  union { uint4 u[2]; v16bf v; } un;
  un.u[0] = *(const uint4*)q;
  un.u[1] = *(const uint4*)(q + 16);
  return un.v;
}

__device__ __forceinline__ v8f wmma_bf16(v16bf a, v16bf b, v8f c) {
  return __builtin_amdgcn_wmma_f32_16x16x32_bf16(false, a, false, b, (short)0, c, false, false);
}

__device__ __forceinline__ float silu_f(float x) { return x / (1.f + __expf(-x)); }

// ---------------------------------------------------------------------------
// CDNA5 async global->LDS copy (16B per lane), ASYNCcnt-tracked.
// ---------------------------------------------------------------------------
__device__ __forceinline__ void async_copy16(const bf16* lds_dst, const bf16* gsrc) {
  unsigned lds_off = (unsigned)(uintptr_t)lds_dst;  // low 32 bits = LDS byte offset
  unsigned long long ga = (unsigned long long)(uintptr_t)gsrc;
  asm volatile("global_load_async_to_lds_b128 %0, %1, off"
               :: "v"(lds_off), "v"(ga) : "memory");
}
__device__ __forceinline__ void wait_async0() {
  asm volatile("s_wait_asynccnt 0" ::: "memory");
}

// ---------------------------------------------------------------------------
// Weight / activation staging kernels
// ---------------------------------------------------------------------------
__global__ void k_cvt_pad(const float* __restrict__ src, bf16* __restrict__ dst,
                          int src_rows, int src_cols, int dst_cols, long total) {
  long i = (long)blockIdx.x * blockDim.x + threadIdx.x;
  if (i >= total) return;
  int r = (int)(i / dst_cols), c = (int)(i % dst_cols);
  float v = (r < src_rows && c < src_cols) ? src[(long)r * src_cols + c] : 0.f;
  dst[i] = (bf16)v;
}

// conv weights (D,C,T) -> (T,D,C) bf16
__global__ void k_stage_conv(const float* __restrict__ w, bf16* __restrict__ dst,
                             int Dout, int Cin, int T, long total) {
  long i = (long)blockIdx.x * blockDim.x + threadIdx.x;
  if (i >= total) return;
  int t = (int)(i / ((long)Dout * Cin));
  long rem = i % ((long)Dout * Cin);
  int d = (int)(rem / Cin), c = (int)(rem % Cin);
  dst[i] = (bf16)w[((long)d * Cin + c) * T + t];
}

// ---------------------------------------------------------------------------
// GroupNorm (num_groups=1) stats + apply(+SiLU) into zero-padded bf16 [B][C][L+2]
// ---------------------------------------------------------------------------
__global__ __launch_bounds__(256) void k_gn_stats(const float* __restrict__ x,
                                                  float* __restrict__ stats, int CL,
                                                  float eps) {
  long base = (long)blockIdx.x * CL;
  float s = 0.f, ss = 0.f;
  for (int i = threadIdx.x; i < CL; i += 256) {
    float v = x[base + i];
    s += v; ss += v * v;
  }
  __shared__ float sh[256], sh2[256];
  sh[threadIdx.x] = s; sh2[threadIdx.x] = ss;
  __syncthreads();
  for (int o = 128; o > 0; o >>= 1) {
    if ((int)threadIdx.x < o) { sh[threadIdx.x] += sh[threadIdx.x + o]; sh2[threadIdx.x] += sh2[threadIdx.x + o]; }
    __syncthreads();
  }
  if (threadIdx.x == 0) {
    float mean = sh[0] / CL;
    float var  = sh2[0] / CL - mean * mean;
    stats[blockIdx.x * 2 + 0] = mean;
    stats[blockIdx.x * 2 + 1] = rsqrtf(var + eps);
  }
}

__global__ void k_gn_silu_pad(const float* __restrict__ x, const float* __restrict__ stats,
                              const float* __restrict__ w, const float* __restrict__ b,
                              bf16* __restrict__ dst, int C, long total) {
  long i = (long)blockIdx.x * blockDim.x + threadIdx.x;
  if (i >= total) return;
  int bi = (int)(i / ((long)C * Ln));
  long rem = i % ((long)C * Ln);
  int c = (int)(rem / Ln), l = (int)(rem % Ln);
  float mean = stats[bi * 2], rstd = stats[bi * 2 + 1];
  float v = (x[i] - mean) * rstd * w[c] + b[c];
  v = silu_f(v);
  long drow = ((long)bi * C + c) * (Ln + 2);
  dst[drow + 1 + l] = (bf16)v;
  if (l == 0)      dst[drow] = (bf16)0.f;
  if (l == Ln - 1) dst[drow + Ln + 1] = (bf16)0.f;
}

__global__ void k_pad_copy(const float* __restrict__ x, bf16* __restrict__ dst,
                           int C, long total) {
  long i = (long)blockIdx.x * blockDim.x + threadIdx.x;
  if (i >= total) return;
  int bi = (int)(i / ((long)C * Ln));
  long rem = i % ((long)C * Ln);
  int c = (int)(rem / Ln), l = (int)(rem % Ln);
  long drow = ((long)bi * C + c) * (Ln + 2);
  dst[drow + 1 + l] = (bf16)x[i];
  if (l == 0)      dst[drow] = (bf16)0.f;
  if (l == Ln - 1) dst[drow + Ln + 1] = (bf16)0.f;
}

// ---------------------------------------------------------------------------
// Conv1d as tap-accumulated WMMA GEMM.
// Block = 8 waves, tile 128(d) x 16(l).  The activation halo tile (18 cols x Cin)
// is staged TRANSPOSED in LDS so B-fragments are k-contiguous (ds_load_b128).
// Wt: [T][Dout][Cin] bf16,  Apad: [B][Cin][L+2] bf16,  out: [B][Dout][L] f32
// ---------------------------------------------------------------------------
__global__ __launch_bounds__(256) void k_conv_gemm(const bf16* __restrict__ Wt,
                                                   const bf16* __restrict__ Apad,
                                                   const float* __restrict__ bias,
                                                   float* __restrict__ out,
                                                   int Cin, int Dout, int T, int tap_off) {
  __shared__ __align__(16) bf16 At[18 * 1024];   // [l_halo 0..17][Cin]
  int b  = blockIdx.z;
  int d0 = blockIdx.y * 128;
  int l0 = blockIdx.x * 16;
  int wave = threadIdx.x >> 5;

  const bf16* Ab = Apad + (long)b * Cin * (Ln + 2);
  int tot = 18 * Cin;
  for (int i = threadIdx.x; i < tot; i += 256) {
    int ci = i / Cin, k = i - ci * Cin;
    At[i] = Ab[(long)k * (Ln + 2) + l0 + ci + tap_off];
  }
  __syncthreads();

  v8f acc = zero8();
  for (int t = 0; t < T; t++) {
    const bf16* Wb = Wt + ((long)t * Dout + d0 + wave * 16) * Cin;
    for (int k0 = 0; k0 < Cin; k0 += 32) {
      // B element (k,n) = At[(t+n)*Cin + k0 + k]  -> kstride 1, nstride Cin
      acc = wmma_bf16(load_a16(Wb + k0, Cin), load_b16(At + t * Cin + k0, Cin), acc);
    }
  }
  int nl = lane_id() & 15, mb = (lane_id() >> 4) << 3;
#pragma unroll
  for (int r = 0; r < 8; r++) {
    int d = d0 + wave * 16 + mb + r;
    out[((long)b * Dout + d) * Ln + l0 + nl] = acc[r] + bias[d];
  }
}

// xp transpose-add: t[(b*L+l)*D+d] = y2[b][d][l] + yskip[b][d][l]
__global__ void k_add_tr(const float* __restrict__ y2, const float* __restrict__ ysk,
                         float* __restrict__ t, long total) {
  long i = (long)blockIdx.x * blockDim.x + threadIdx.x;
  if (i >= total) return;
  int b = (int)(i / ((long)Dn * Ln));
  long rem = i % ((long)Dn * Ln);
  int d = (int)(rem / Ln), l = (int)(rem % Ln);
  t[((long)b * Ln + l) * Dn + d] = y2[i] + ysk[i];
}

// ---------------------------------------------------------------------------
// RMSNorm: one block per row of D=1024; outputs bf16 normalized row
// ---------------------------------------------------------------------------
__global__ __launch_bounds__(256) void k_rms(const float* __restrict__ t,
                                             const float* __restrict__ w,
                                             bf16* __restrict__ out) {
  const float* row = t + (long)blockIdx.x * Dn;
  float ss = 0.f;
  for (int i = threadIdx.x; i < Dn; i += 256) ss += row[i] * row[i];
  __shared__ float sh[256];
  sh[threadIdx.x] = ss;
  __syncthreads();
  for (int o = 128; o > 0; o >>= 1) {
    if ((int)threadIdx.x < o) sh[threadIdx.x] += sh[threadIdx.x + o];
    __syncthreads();
  }
  float rs = rsqrtf(sh[0] / Dn + 1e-6f);
  for (int i = threadIdx.x; i < Dn; i += 256)
    out[(long)blockIdx.x * Dn + i] = (bf16)(row[i] * rs * w[i]);
}

// ---------------------------------------------------------------------------
// Generic bf16 GEMM:  out[M,N] = act[M,K] @ W[N,K]^T  (+bias)(+resid)(swiglu)
// block = 256 threads (8 waves), tile 128(M) x 64(N), K step 32.
// Tiles staged via GLOBAL_LOAD_ASYNC_TO_LDS_B128 (ASYNCcnt).
// ---------------------------------------------------------------------------
__global__ __launch_bounds__(256) void k_gemm(const bf16* __restrict__ A, int lda,
                                              const bf16* __restrict__ W, int ldw,
                                              const float* __restrict__ bias,
                                              const float* __restrict__ resid,
                                              const float* __restrict__ aux,
                                              float* __restrict__ outf,
                                              bf16* __restrict__ outh,
                                              int ldo, int K, int swiglu) {
  __shared__ __align__(16) bf16 As[128 * 32];
  __shared__ __align__(16) bf16 Bs[64 * 32];
  int m0 = blockIdx.y * 128, n0 = blockIdx.x * 64;
  int wave = threadIdx.x >> 5;
  v8f acc[4];
#pragma unroll
  for (int j = 0; j < 4; j++) acc[j] = zero8();

  for (int k0 = 0; k0 < K; k0 += 32) {
    __syncthreads();
#pragma unroll
    for (int i = threadIdx.x; i < 512; i += 256) {
      int r = i >> 2, q = i & 3;
      async_copy16(As + r * 32 + q * 8, A + (long)(m0 + r) * lda + k0 + q * 8);
    }
    {
      int r = threadIdx.x >> 2, q = threadIdx.x & 3;
      async_copy16(Bs + r * 32 + q * 8, W + (long)(n0 + r) * ldw + k0 + q * 8);
    }
    wait_async0();
    __syncthreads();
    v16bf af = load_a16(As + wave * 16 * 32, 32);
#pragma unroll
    for (int j = 0; j < 4; j++)
      acc[j] = wmma_bf16(af, load_b16(Bs + j * 16 * 32, 32), acc[j]);
  }

  int nl = lane_id() & 15, mb = (lane_id() >> 4) << 3;
#pragma unroll
  for (int j = 0; j < 4; j++) {
#pragma unroll
    for (int r = 0; r < 8; r++) {
      int gm = m0 + wave * 16 + mb + r;
      int gn = n0 + j * 16 + nl;
      float v = acc[j][r];
      if (bias) v += bias[gn];
      if (swiglu) { float x1 = aux[(long)gm * ldo + gn]; v = silu_f(x1) * v; }
      if (resid) v += resid[(long)gm * ldo + gn];
      long o = (long)gm * ldo + gn;
      if (outf) outf[o] = v;
      if (outh) outh[o] = (bf16)v;
    }
  }
}

// ---------------------------------------------------------------------------
// RoPE + head split: qkv f32 [B*L][3072] -> Q/K bf16 [B][H][L][64],
// V bf16 TRANSPOSED [B][H][64][L] (so P@V B-fragments are k-contiguous).
// ---------------------------------------------------------------------------
__global__ void k_rope(const float* __restrict__ qkv, const float* __restrict__ fc,
                       bf16* __restrict__ Q, bf16* __restrict__ K, bf16* __restrict__ V,
                       long total) {
  long i = (long)blockIdx.x * blockDim.x + threadIdx.x;
  if (i >= total) return;
  int b = (int)(i / ((long)Ln * Hn * 32));
  long rem = i % ((long)Ln * Hn * 32);
  int l = (int)(rem / (Hn * 32));
  int rem2 = (int)(rem % (Hn * 32));
  int h = rem2 / 32, p = rem2 % 32;
  const float* base = qkv + (long)(b * Ln + l) * QKVN;
  float c = fc[(l * 32 + p) * 2], s = fc[(l * 32 + p) * 2 + 1];
  int col = h * 64 + 2 * p;
  float q0 = base[col], q1 = base[col + 1];
  float k0 = base[Dn + col], k1 = base[Dn + col + 1];
  float v0 = base[2 * Dn + col], v1 = base[2 * Dn + col + 1];
  long hb = (((long)b * Hn + h) * Ln + l) * 64 + 2 * p;
  Q[hb] = (bf16)(q0 * c - q1 * s); Q[hb + 1] = (bf16)(q1 * c + q0 * s);
  K[hb] = (bf16)(k0 * c - k1 * s); K[hb + 1] = (bf16)(k1 * c + k0 * s);
  long vb = (((long)b * Hn + h) * 64 + 2 * p) * Ln + l;
  V[vb] = (bf16)v0; V[vb + Ln] = (bf16)v1;
}

// ---------------------------------------------------------------------------
// Flash attention: block = 4 waves, 64 q-rows, loop over 16 k-tiles of 64.
// Q/K bf16 [B][H][L][64], V bf16 [B][H][64][L]; output o bf16 [B*L][1024]
// ---------------------------------------------------------------------------
__global__ __launch_bounds__(128) void k_flash(const bf16* __restrict__ Q,
                                               const bf16* __restrict__ K,
                                               const bf16* __restrict__ V,
                                               bf16* __restrict__ O, float scale) {
  int b = blockIdx.z, h = blockIdx.y, q0 = blockIdx.x * 64;
  int wave = threadIdx.x >> 5, lane = lane_id();
  __shared__ __align__(16) bf16 Kt[64 * 64];   // [krow][dh]
  __shared__ __align__(16) bf16 Vt[64 * 64];   // [dh][krow]
  __shared__ float Sld[4][16 * 64];
  __shared__ __align__(16) bf16 Pld[4][16 * 64];
  __shared__ float facs[4][16], rowm[4][16], rowl[4][16];

  const bf16* Qh  = Q + (((long)b * Hn + h) * Ln + q0) * 64;
  const bf16* Kh  = K + ((long)b * Hn + h) * Ln * 64;
  const bf16* VhT = V + ((long)b * Hn + h) * 64 * Ln;

  v16bf qa0 = load_a16(Qh + (long)wave * 16 * 64, 64);
  v16bf qa1 = load_a16(Qh + (long)wave * 16 * 64 + 32, 64);

  v8f acc[4];
#pragma unroll
  for (int j = 0; j < 4; j++) acc[j] = zero8();
  if (lane < 16) { rowm[wave][lane] = -1e30f; rowl[wave][lane] = 0.f; }

  for (int kt = 0; kt < Ln / 64; kt++) {
    __syncthreads();
    {
      const uint4* Ks = (const uint4*)(Kh + (long)kt * 64 * 64);
      uint4* Kd = (uint4*)Kt;
      uint4* Vd = (uint4*)Vt;
      for (int i = threadIdx.x; i < 512; i += 128) {
        Kd[i] = Ks[i];
        int row = i >> 3, c = i & 7;   // Vt row = dh, 8 uint4 per row of 64
        Vd[i] = *((const uint4*)(VhT + (long)row * Ln + kt * 64) + c);
      }
    }
    __syncthreads();

    // S = Q K^T  (16 x 64 per wave);  B elem (k=dh, n=krow) = Kt[krow*64+dh]
    v8f s[4];
#pragma unroll
    for (int j = 0; j < 4; j++) s[j] = zero8();
#pragma unroll
    for (int j = 0; j < 4; j++) {
      s[j] = wmma_bf16(qa0, load_b16(Kt + j * 16 * 64, 64), s[j]);
      s[j] = wmma_bf16(qa1, load_b16(Kt + j * 16 * 64 + 32, 64), s[j]);
    }
    int nl = lane & 15, mb = (lane >> 4) << 3;
#pragma unroll
    for (int j = 0; j < 4; j++)
#pragma unroll
      for (int r = 0; r < 8; r++)
        Sld[wave][(mb + r) * 64 + j * 16 + nl] = s[j][r] * scale;

    // online softmax: 2 lanes per row, 32 cols each
    int row = lane >> 1, half = lane & 1;
    float* Srow = &Sld[wave][row * 64 + half * 32];
    float mx = -1e30f;
#pragma unroll
    for (int c = 0; c < 32; c++) mx = fmaxf(mx, Srow[c]);
    mx = fmaxf(mx, __shfl_xor(mx, 1, 32));
    float mold = rowm[wave][row];
    float mnew = fmaxf(mold, mx);
    float f = __expf(mold - mnew);
    float sum = 0.f;
#pragma unroll
    for (int c = 0; c < 32; c++) {
      float p = __expf(Srow[c] - mnew);
      sum += p;
      Pld[wave][row * 64 + half * 32 + c] = (bf16)p;
    }
    sum += __shfl_xor(sum, 1, 32);
    if (half == 0) {
      rowm[wave][row] = mnew;
      rowl[wave][row] = rowl[wave][row] * f + sum;
      facs[wave][row] = f;
    }

    // rescale accumulators and add P @ V
#pragma unroll
    for (int j = 0; j < 4; j++)
#pragma unroll
      for (int r = 0; r < 8; r++) acc[j][r] *= facs[wave][mb + r];

    v16bf pa0 = load_a16(&Pld[wave][0], 64);
    v16bf pa1 = load_a16(&Pld[wave][32], 64);
#pragma unroll
    for (int j = 0; j < 4; j++) {
      // B elem (k=krow, n=dh) = Vt[(j*16+n)*64 + k]
      acc[j] = wmma_bf16(pa0, load_b16(Vt + (j * 16) * 64, 64), acc[j]);
      acc[j] = wmma_bf16(pa1, load_b16(Vt + (j * 16) * 64 + 32, 64), acc[j]);
    }
  }

  int nl = lane & 15, mb = (lane >> 4) << 3;
#pragma unroll
  for (int j = 0; j < 4; j++)
#pragma unroll
    for (int r = 0; r < 8; r++) {
      int gq = q0 + wave * 16 + mb + r;
      float inv = 1.f / rowl[wave][mb + r];
      O[((long)b * Ln + gq) * Dn + h * 64 + j * 16 + nl] = (bf16)(acc[j][r] * inv);
    }
}

// ---------------------------------------------------------------------------
// Cross-attention with top-k softmax.  block = 128 threads (4 waves),
// 16 q-rows x 512 style keys.  Scores via WMMA into LDS, then per-row top-k.
// ---------------------------------------------------------------------------
__global__ __launch_bounds__(128) void k_xattn(const bf16* __restrict__ Q2,
                                               const bf16* __restrict__ K2,
                                               const float* __restrict__ styles,
                                               const int* __restrict__ topk_p,
                                               float* __restrict__ out, float scale) {
  int b = blockIdx.z, h = blockIdx.y, q0 = blockIdx.x * 16;
  int wave = threadIdx.x >> 5, lane = lane_id();
  __shared__ float S[16 * 512];
  __shared__ float cand[16][8][8];
  __shared__ float cut[16], mxs[16], sums[16];

  const bf16* Qb = Q2 + (long)(b * Ln + q0) * Dn + h * 64;
  v16bf a0 = load_a16(Qb, Dn);
  v16bf a1 = load_a16(Qb + 32, Dn);

  for (int j = 0; j < 8; j++) {
    int n0 = (wave * 8 + j) * 16;
    const bf16* Kb = K2 + (long)(b * Sn + n0) * Dn + h * 64;
    v8f s = zero8();
    s = wmma_bf16(a0, load_b16(Kb, Dn), s);
    s = wmma_bf16(a1, load_b16(Kb + 32, Dn), s);
    int nl = lane & 15, mb = (lane >> 4) << 3;
#pragma unroll
    for (int r = 0; r < 8; r++) S[(mb + r) * 512 + n0 + nl] = s[r] * scale;
  }
  __syncthreads();

  int Kk = *topk_p; if (Kk < 1) Kk = 1; if (Kk > 8) Kk = 8;
  int row = threadIdx.x >> 3, sub = threadIdx.x & 7;

  // local top-8 over 64 scores
  float loc[8];
#pragma unroll
  for (int i = 0; i < 8; i++) loc[i] = -1e30f;
  const float* Sr = S + row * 512 + sub * 64;
  for (int c = 0; c < 64; c++) {
    float v = Sr[c];
    if (v > loc[7]) {
      loc[7] = v;
      for (int i = 7; i > 0 && loc[i] > loc[i - 1]; --i) {
        float tmp = loc[i]; loc[i] = loc[i - 1]; loc[i - 1] = tmp;
      }
    }
  }
#pragma unroll
  for (int i = 0; i < 8; i++) cand[row][sub][i] = loc[i];
  __syncthreads();

  if (sub == 0) {
    float arr[64];
    for (int t = 0; t < 8; t++)
      for (int i = 0; i < 8; i++) arr[t * 8 + i] = cand[row][t][i];
    float best = -1e30f, cutoff = -1e30f;
    for (int sel = 0; sel < Kk; sel++) {
      int bi = 0; float bv = -1e31f;
      for (int i2 = 0; i2 < 64; i2++)
        if (arr[i2] > bv) { bv = arr[i2]; bi = i2; }
      if (sel == 0) best = bv;
      cutoff = bv; arr[bi] = -1e32f;
    }
    cut[row] = cutoff; mxs[row] = best; sums[row] = 0.f;
  }
  __syncthreads();

  float part = 0.f;
  const float* Sr2 = S + row * 512;
  for (int c = sub; c < 512; c += 8) {
    float v = Sr2[c];
    if (v >= cut[row]) part += __expf(v - mxs[row]);
  }
  atomicAdd(&sums[row], part);
  __syncthreads();

  float inv = 1.f / sums[row];
  float accv[8];
#pragma unroll
  for (int i = 0; i < 8; i++) accv[i] = 0.f;
  const float* Vb = styles + (long)b * Sn * Dn + h * 64 + sub * 8;
  for (int s2 = 0; s2 < 512; s2++) {
    float v = Sr2[s2];
    if (v >= cut[row]) {
      float w = __expf(v - mxs[row]) * inv;
      const float* vr = Vb + (long)s2 * Dn;
#pragma unroll
      for (int i = 0; i < 8; i++) accv[i] += w * vr[i];
    }
  }
  float* ob = out + (long)(b * Ln + q0 + row) * Dn + h * 64 + sub * 8;
#pragma unroll
  for (int i = 0; i < 8; i++) ob[i] = accv[i];
}

// ---------------------------------------------------------------------------
// Host orchestration
// ---------------------------------------------------------------------------
static inline int divup_l(long a, int b) { return (int)((a + b - 1) / b); }

extern "C" void kernel_launch(void* const* d_in, const int* in_sizes, int n_in,
                              void* d_out, int out_size, void* d_ws, size_t ws_size,
                              hipStream_t stream) {
  (void)in_sizes; (void)n_in; (void)out_size; (void)ws_size;
  const float* x        = (const float*)d_in[0];
  const float* styles   = (const float*)d_in[1];
  const float* freqs    = (const float*)d_in[2];
  const float* gn1_w    = (const float*)d_in[3];
  const float* gn1_b    = (const float*)d_in[4];
  const float* conv1_w  = (const float*)d_in[5];
  const float* conv1_b  = (const float*)d_in[6];
  const float* gn2_w    = (const float*)d_in[7];
  const float* gn2_b    = (const float*)d_in[8];
  const float* conv2_w  = (const float*)d_in[9];
  const float* conv2_b  = (const float*)d_in[10];
  const float* skip_w   = (const float*)d_in[11];
  const float* skip_b   = (const float*)d_in[12];
  const float* norm1_w  = (const float*)d_in[13];
  const float* qkv_w    = (const float*)d_in[14];
  const float* proj_w   = (const float*)d_in[15];
  const float* proj_b   = (const float*)d_in[16];
  const float* norm2_w  = (const float*)d_in[17];
  const float* w12_w    = (const float*)d_in[18];
  const float* w3_w     = (const float*)d_in[19];
  const float* q_w      = (const float*)d_in[20];
  const float* k_w      = (const float*)d_in[21];
  const int*   top_k    = (const int*)d_in[22];
  float* out = (float*)d_out;

  char* wsb = (char*)d_ws;
  size_t off = 0;
  auto alloc = [&](size_t bytes) -> void* {
    void* p = wsb + off;
    off = (off + bytes + 255) & ~(size_t)255;
    return p;
  };

  float* gn1_st = (float*)alloc(Bn * 2 * 4);
  float* gn2_st = (float*)alloc(Bn * 2 * 4);
  bf16* xpad   = (bf16*)alloc((size_t)Bn * CINn * (Ln + 2) * 2);
  bf16* a1pad  = (bf16*)alloc((size_t)Bn * CINn * (Ln + 2) * 2);
  bf16* wtap1  = (bf16*)alloc((size_t)3 * Dn * CINn * 2);
  bf16* wtap2  = (bf16*)alloc((size_t)3 * Dn * Dn * 2);
  bf16* wskip  = (bf16*)alloc((size_t)Dn * CINn * 2);
  float* y1    = (float*)alloc((size_t)Bn * Dn * Ln * 4);   // aliased as t2 later
  bf16* a2pad  = (bf16*)alloc((size_t)Bn * Dn * (Ln + 2) * 2);
  float* y2    = (float*)alloc((size_t)Bn * Dn * Ln * 4);   // aliased as t3 later
  float* yskip = (float*)alloc((size_t)Bn * Dn * Ln * 4);
  float* t     = (float*)alloc((size_t)Mrow * Dn * 4);
  bf16* n_h    = (bf16*)alloc((size_t)Mrow * Dn * 2);
  bf16* qkvw_h = (bf16*)alloc((size_t)QKVN * Dn * 2);
  float* qkv   = (float*)alloc((size_t)Mrow * QKVN * 4);    // aliased as x1 later
  bf16* q_hm   = (bf16*)alloc((size_t)Mrow * Dn * 2);
  bf16* k_hm   = (bf16*)alloc((size_t)Mrow * Dn * 2);
  bf16* v_hm   = (bf16*)alloc((size_t)Mrow * Dn * 2);
  bf16* o_h    = (bf16*)alloc((size_t)Mrow * Dn * 2);
  bf16* projw_h= (bf16*)alloc((size_t)Dn * Dn * 2);
  bf16* m_h    = (bf16*)alloc((size_t)Mrow * Dn * 2);
  bf16* w12a_h = (bf16*)alloc((size_t)HIDP * Dn * 2);
  bf16* w12b_h = (bf16*)alloc((size_t)HIDP * Dn * 2);
  bf16* g_h    = (bf16*)alloc((size_t)Mrow * HIDP * 2);
  bf16* w3w_h  = (bf16*)alloc((size_t)Dn * HIDP * 2);
  bf16* t3_h   = (bf16*)alloc((size_t)Mrow * Dn * 2);
  bf16* qw_h   = (bf16*)alloc((size_t)Dn * Dn * 2);
  bf16* kw_h   = (bf16*)alloc((size_t)Dn * Dn * 2);
  bf16* q2_h   = (bf16*)alloc((size_t)Mrow * Dn * 2);
  bf16* sty_h  = (bf16*)alloc((size_t)Msty * Dn * 2);
  bf16* k2_h   = (bf16*)alloc((size_t)Msty * Dn * 2);
  float* t2 = y1;     // y1 dead after gn2 apply
  float* t3 = y2;     // y2 dead after transpose-add
  float* x1 = qkv;    // qkv dead after rope

  const float scale = 0.125f;  // DH^-0.5

  // --- stage weights / styles to bf16 (with padding) ---
  {
    long tt;
    tt = (long)QKVN * Dn;
    k_cvt_pad<<<divup_l(tt, 256), 256, 0, stream>>>(qkv_w, qkvw_h, QKVN, Dn, Dn, tt);
    tt = (long)Dn * Dn;
    k_cvt_pad<<<divup_l(tt, 256), 256, 0, stream>>>(proj_w, projw_h, Dn, Dn, Dn, tt);
    k_cvt_pad<<<divup_l(tt, 256), 256, 0, stream>>>(q_w, qw_h, Dn, Dn, Dn, tt);
    k_cvt_pad<<<divup_l(tt, 256), 256, 0, stream>>>(k_w, kw_h, Dn, Dn, Dn, tt);
    tt = (long)HIDP * Dn;
    k_cvt_pad<<<divup_l(tt, 256), 256, 0, stream>>>(w12_w, w12a_h, HIDn, Dn, Dn, tt);
    k_cvt_pad<<<divup_l(tt, 256), 256, 0, stream>>>(w12_w + (long)HIDn * Dn, w12b_h, HIDn, Dn, Dn, tt);
    tt = (long)Dn * HIDP;
    k_cvt_pad<<<divup_l(tt, 256), 256, 0, stream>>>(w3_w, w3w_h, Dn, HIDn, HIDP, tt);
    tt = (long)Msty * Dn;
    k_cvt_pad<<<divup_l(tt, 256), 256, 0, stream>>>(styles, sty_h, Msty, Dn, Dn, tt);
    tt = (long)3 * Dn * CINn;
    k_stage_conv<<<divup_l(tt, 256), 256, 0, stream>>>(conv1_w, wtap1, Dn, CINn, 3, tt);
    tt = (long)3 * Dn * Dn;
    k_stage_conv<<<divup_l(tt, 256), 256, 0, stream>>>(conv2_w, wtap2, Dn, Dn, 3, tt);
    tt = (long)Dn * CINn;
    k_stage_conv<<<divup_l(tt, 256), 256, 0, stream>>>(skip_w, wskip, Dn, CINn, 1, tt);
  }

  // --- conv stack ---
  k_gn_stats<<<Bn, 256, 0, stream>>>(x, gn1_st, CINn * Ln, 1e-5f);
  {
    long tt = (long)Bn * CINn * Ln;
    k_gn_silu_pad<<<divup_l(tt, 256), 256, 0, stream>>>(x, gn1_st, gn1_w, gn1_b, a1pad, CINn, tt);
    k_pad_copy<<<divup_l(tt, 256), 256, 0, stream>>>(x, xpad, CINn, tt);
  }
  k_conv_gemm<<<dim3(Ln / 16, Dn / 128, Bn), 256, 0, stream>>>(wtap1, a1pad, conv1_b, y1, CINn, Dn, 3, 0);
  k_gn_stats<<<Bn, 256, 0, stream>>>(y1, gn2_st, Dn * Ln, 1e-5f);
  {
    long tt = (long)Bn * Dn * Ln;
    k_gn_silu_pad<<<divup_l(tt, 256), 256, 0, stream>>>(y1, gn2_st, gn2_w, gn2_b, a2pad, Dn, tt);
  }
  k_conv_gemm<<<dim3(Ln / 16, Dn / 128, Bn), 256, 0, stream>>>(wtap2, a2pad, conv2_b, y2, Dn, Dn, 3, 0);
  k_conv_gemm<<<dim3(Ln / 16, Dn / 128, Bn), 256, 0, stream>>>(wskip, xpad, skip_b, yskip, CINn, Dn, 1, 1);
  {
    long tt = (long)Bn * Dn * Ln;
    k_add_tr<<<divup_l(tt, 256), 256, 0, stream>>>(y2, yskip, t, tt);
  }

  // --- self-attention block ---
  k_rms<<<Mrow, 256, 0, stream>>>(t, norm1_w, n_h);
  k_gemm<<<dim3(QKVN / 64, Mrow / 128), 256, 0, stream>>>(
      n_h, Dn, qkvw_h, Dn, nullptr, nullptr, nullptr, qkv, nullptr, QKVN, Dn, 0);
  {
    long tt = (long)Bn * Ln * Hn * 32;
    k_rope<<<divup_l(tt, 256), 256, 0, stream>>>(qkv, freqs, q_hm, k_hm, v_hm, tt);
  }
  k_flash<<<dim3(Ln / 64, Hn, Bn), 128, 0, stream>>>(q_hm, k_hm, v_hm, o_h, scale);
  k_gemm<<<dim3(Dn / 64, Mrow / 128), 256, 0, stream>>>(
      o_h, Dn, projw_h, Dn, proj_b, t, nullptr, t2, nullptr, Dn, Dn, 0);

  // --- SwiGLU MLP ---
  k_rms<<<Mrow, 256, 0, stream>>>(t2, norm2_w, m_h);
  k_gemm<<<dim3(HIDP / 64, Mrow / 128), 256, 0, stream>>>(
      m_h, Dn, w12a_h, Dn, nullptr, nullptr, nullptr, x1, nullptr, HIDP, Dn, 0);
  k_gemm<<<dim3(HIDP / 64, Mrow / 128), 256, 0, stream>>>(
      m_h, Dn, w12b_h, Dn, nullptr, nullptr, x1, nullptr, g_h, HIDP, Dn, 1);
  k_gemm<<<dim3(Dn / 64, Mrow / 128), 256, 0, stream>>>(
      g_h, HIDP, w3w_h, HIDP, nullptr, t2, nullptr, t3, t3_h, Dn, HIDP, 0);

  // --- style cross-attention ---
  k_gemm<<<dim3(Dn / 64, Mrow / 128), 256, 0, stream>>>(
      t3_h, Dn, qw_h, Dn, nullptr, nullptr, nullptr, nullptr, q2_h, Dn, Dn, 0);
  k_gemm<<<dim3(Dn / 64, Msty / 128), 256, 0, stream>>>(
      sty_h, Dn, kw_h, Dn, nullptr, nullptr, nullptr, nullptr, k2_h, Dn, Dn, 0);
  k_xattn<<<dim3(Ln / 16, Hn, Bn), 128, 0, stream>>>(q2_h, k2_h, styles, top_k, out, scale);
  (void)t3;
}
